// SINConv_4372276707359
// MI455X (gfx1250) — matrix operand order repaired and checked by reading, not compiled, not run on hardware
//
#include <hip/hip_runtime.h>
#include <cstdint>
#include <cstddef>

// ---------------------------------------------------------------------------
// MI455X (gfx1250, wave32) SINConv-style GNN forward.
// GEMMs routed through v_wmma_f32_16x16x32_bf16 (fp32 accumulate).
// ~32 GFLOP total vs ~540 MB HBM traffic -> memory-bound (~25us) if matrix
// path is used; fp32 VALU would be compute-bound, so bf16 WMMA is the choice.
// ---------------------------------------------------------------------------

typedef __attribute__((ext_vector_type(16))) __bf16 bf16x16;
typedef __attribute__((ext_vector_type(8)))  float  f32x8;

union AFrag {
    bf16x16      v;
    uint4        q[2];
    unsigned int u[8];
};

__device__ __forceinline__ unsigned int pk_bf16(float a, float b) {
    union { __bf16 h[2]; unsigned int u; } t;
    t.h[0] = (__bf16)a;
    t.h[1] = (__bf16)b;
    return t.u;
}

__device__ __forceinline__ float relu_f(float v) { return v > 0.0f ? v : 0.0f; }

constexpr int Dv = 64;     // feature dim
constexpr int K2 = 128;    // concat dim
constexpr int AT = 136;    // LDS stride for 128-wide tiles (pad, mult of 8)
constexpr int ZS = 72;     // LDS stride for 64-wide tiles  (pad, mult of 8)

// ---------------------------------------------------------------------------
// Prep: weights fp32 row-major (K,N) -> bf16 transposed (N,K)
// ---------------------------------------------------------------------------
__global__ void prep_weights_kernel(const float* __restrict__ Wup,
                                    const float* __restrict__ Wdn,
                                    const float* __restrict__ W1,
                                    const float* __restrict__ W2,
                                    __bf16* __restrict__ WupT,
                                    __bf16* __restrict__ WdnT,
                                    __bf16* __restrict__ W1T,
                                    __bf16* __restrict__ W2T) {
    int i = blockIdx.x * blockDim.x + threadIdx.x;
    if (i < K2 * Dv) {
        int k = i >> 6, n = i & 63;
        WupT[n * K2 + k] = (__bf16)Wup[i];
        WdnT[n * K2 + k] = (__bf16)Wdn[i];
        if (k < Dv) {
            W1T[n * Dv + k] = (__bf16)W1[i];
            W2T[n * Dv + k] = (__bf16)W2[i];
        }
    }
}

// Per level: x fp32 -> bf16 copy (halves gather bytes) and acc = 2*(1+eps)*x
__global__ void prep_level_kernel(const float* __restrict__ x,
                                  __bf16* __restrict__ xbf,
                                  float* __restrict__ acc, int n) {
    int i = blockIdx.x * blockDim.x + threadIdx.x;
    if (i < n) {
        float v = x[i];
        xbf[i] = (__bf16)v;
        acc[i] = 2.0f * v;   // EPS == 0
    }
}

// ---------------------------------------------------------------------------
// Edge MLP: m = relu([x[src] | attr] @ W + b); atomic scatter-add into acc.
// Block = 128 threads = 4 waves; each wave owns a 16-edge x 64-out tile.
// ---------------------------------------------------------------------------
__global__ __launch_bounds__(128) void edge_mlp_kernel(
    const __bf16* __restrict__ xbf,   // (N, 64) bf16
    const int*    __restrict__ idx,   // (2, E): [0]=dst, [1]=src
    const float*  __restrict__ attr,  // (E, 64) fp32
    const __bf16* __restrict__ Wt,    // (64, 128) bf16, transposed
    const float*  __restrict__ bias,  // (64,)
    float*        __restrict__ accum, // (N, 64) fp32
    int E_) {
    __shared__ __align__(16) __bf16 sA[4][16 * AT];
    __shared__ __align__(16) __bf16 sW[Dv * AT];
    __shared__ int sDst[4][16];

    const int tid  = threadIdx.x;
    const int wave = tid >> 5;
    const int lane = tid & 31;

    // Cooperative load of Wt (64x128 bf16) into padded LDS.
    for (int i = tid; i < Dv * (K2 / 8); i += 128) {
        int n = i >> 4;
        int c = (i & 15) * 8;
        *(uint4*)&sW[n * AT + c] = *(const uint4*)&Wt[n * K2 + c];
    }
    __syncthreads();

    const int e0  = (blockIdx.x * 4 + wave) * 16;
    int cnt = E_ - e0;
    if (cnt > 16) cnt = 16;
    if (cnt <= 0) return;

    const int r     = lane & 15;
    const int half  = lane >> 4;
    const int base8 = half ? 8 : 0;   // K-offset (and M-offset) for hi lanes
    const int nlo   = lane & 15;

    // ---- stage A tile: row r = edge e0+r; cols 0..63 = x[src], 64..127 = attr
    {
        int e     = e0 + r;
        int esafe = (r < cnt) ? e : e0;
        if (half == 0) {
            int src = idx[E_ + esafe];
            const uint4* xp = (const uint4*)(xbf + (size_t)src * Dv);
            uint4* dp = (uint4*)&sA[wave][r * AT];
#pragma unroll
            for (int j = 0; j < 8; ++j) dp[j] = xp[j];
            sDst[wave][r] = idx[esafe];
        } else {
            const float4* ap = (const float4*)(attr + (size_t)esafe * Dv);
            unsigned int* dp = (unsigned int*)&sA[wave][r * AT + Dv];
#pragma unroll
            for (int j = 0; j < 16; ++j) {
                float4 f = ap[j];
                dp[j * 2 + 0] = pk_bf16(f.x, f.y);
                dp[j * 2 + 1] = pk_bf16(f.z, f.w);
            }
        }
    }
    asm volatile("s_wait_dscnt 0" ::: "memory");  // intra-wave LDS RAW

    float bias4[4];
#pragma unroll
    for (int t = 0; t < 4; ++t) bias4[t] = bias[t * 16 + nlo];

    f32x8 accv[4];
#pragma unroll
    for (int t = 0; t < 4; ++t)
#pragma unroll
        for (int j = 0; j < 8; ++j) accv[t][j] = 0.0f;

    const __bf16* aRow = &sA[wave][r * AT];
#pragma unroll
    for (int kb = 0; kb < 4; ++kb) {
        AFrag a;
        a.q[0] = *(const uint4*)&aRow[kb * 32 + base8];
        a.q[1] = *(const uint4*)&aRow[kb * 32 + base8 + 16];
#pragma unroll
        for (int t = 0; t < 4; ++t) {
            AFrag b;
            const __bf16* wRow =
                &sW[(t * 16 + nlo) * AT + kb * 32 + (half ? 8 : 0)];
            b.q[0] = *(const uint4*)&wRow[0];
            b.q[1] = *(const uint4*)&wRow[16];
            accv[t] = __builtin_amdgcn_wmma_f32_16x16x32_bf16(
                false, a.v, false, b.v, (short)0, accv[t], false, false);
        }
    }

    // Epilogue: D(m,n): VGPR v -> m = v + (lane>=16 ? 8 : 0), n = t*16 + nlo
#pragma unroll
    for (int t = 0; t < 4; ++t) {
#pragma unroll
        for (int v = 0; v < 8; ++v) {
            int m = v + base8;
            if (m < cnt) {
                float val = relu_f(accv[t][v] + bias4[t]);
                atomicAdd(&accum[(size_t)sDst[wave][m] * Dv + t * 16 + nlo], val);
            }
        }
    }
}

// ---------------------------------------------------------------------------
// Node MLP: out = relu(relu(h@W1+b1)@W2+b2), h = acc (N,64) fp32.
// ---------------------------------------------------------------------------
__global__ __launch_bounds__(128) void node_mlp_kernel(
    const float*  __restrict__ h,
    const __bf16* __restrict__ W1t, const float* __restrict__ b1,
    const __bf16* __restrict__ W2t, const float* __restrict__ b2,
    float* __restrict__ out, int N_) {
    __shared__ __align__(16) __bf16 sW1[Dv * ZS];
    __shared__ __align__(16) __bf16 sW2[Dv * ZS];
    __shared__ __align__(16) __bf16 sZ[4][16 * ZS];

    const int tid  = threadIdx.x;
    const int wave = tid >> 5;
    const int lane = tid & 31;

    for (int i = tid; i < Dv * 8; i += 128) {
        int n = i >> 3;
        int c = (i & 7) * 8;
        *(uint4*)&sW1[n * ZS + c] = *(const uint4*)&W1t[n * Dv + c];
        *(uint4*)&sW2[n * ZS + c] = *(const uint4*)&W2t[n * Dv + c];
    }
    __syncthreads();

    const int n0 = (blockIdx.x * 4 + wave) * 16;
    int cnt = N_ - n0;
    if (cnt > 16) cnt = 16;
    if (cnt <= 0) return;

    const int r     = lane & 15;
    const int half  = lane >> 4;
    const int base8 = half ? 8 : 0;
    const int nlo   = lane & 15;

    int node = n0 + r;
    if (r >= cnt) node = n0;  // clamp tail loads to valid memory
    const float* hp = h + (size_t)node * Dv;

    float b1v[4], b2v[4];
#pragma unroll
    for (int t = 0; t < 4; ++t) {
        b1v[t] = b1[t * 16 + nlo];
        b2v[t] = b2[t * 16 + nlo];
    }

    // ---- GEMM1: h @ W1, A loaded straight from global (L2-resident), cvt bf16
    f32x8 accv[4];
#pragma unroll
    for (int t = 0; t < 4; ++t)
#pragma unroll
        for (int j = 0; j < 8; ++j) accv[t][j] = 0.0f;

#pragma unroll
    for (int kb = 0; kb < 2; ++kb) {
        AFrag a;
        const float* hp0 = hp + kb * 32 + base8;
        float4 f0 = *(const float4*)(hp0 + 0);
        float4 f1 = *(const float4*)(hp0 + 4);
        float4 f2 = *(const float4*)(hp0 + 16);
        float4 f3 = *(const float4*)(hp0 + 20);
        a.u[0] = pk_bf16(f0.x, f0.y); a.u[1] = pk_bf16(f0.z, f0.w);
        a.u[2] = pk_bf16(f1.x, f1.y); a.u[3] = pk_bf16(f1.z, f1.w);
        a.u[4] = pk_bf16(f2.x, f2.y); a.u[5] = pk_bf16(f2.z, f2.w);
        a.u[6] = pk_bf16(f3.x, f3.y); a.u[7] = pk_bf16(f3.z, f3.w);
#pragma unroll
        for (int t = 0; t < 4; ++t) {
            AFrag b;
            const __bf16* wRow =
                &sW1[(t * 16 + nlo) * ZS + kb * 32 + (half ? 8 : 0)];
            b.q[0] = *(const uint4*)&wRow[0];
            b.q[1] = *(const uint4*)&wRow[16];
            accv[t] = __builtin_amdgcn_wmma_f32_16x16x32_bf16(
                false, a.v, false, b.v, (short)0, accv[t], false, false);
        }
    }

    // relu + spill intermediate to LDS (D-layout -> row-major bf16)
#pragma unroll
    for (int t = 0; t < 4; ++t) {
#pragma unroll
        for (int v = 0; v < 8; ++v) {
            int m = v + base8;
            float val = relu_f(accv[t][v] + b1v[t]);
            sZ[wave][m * ZS + t * 16 + nlo] = (__bf16)val;
        }
    }
    asm volatile("s_wait_dscnt 0" ::: "memory");

    // ---- GEMM2: z @ W2
#pragma unroll
    for (int t = 0; t < 4; ++t)
#pragma unroll
        for (int j = 0; j < 8; ++j) accv[t][j] = 0.0f;

    const __bf16* zRow = &sZ[wave][r * ZS];
#pragma unroll
    for (int kb = 0; kb < 2; ++kb) {
        AFrag a;
        a.q[0] = *(const uint4*)&zRow[kb * 32 + base8];
        a.q[1] = *(const uint4*)&zRow[kb * 32 + base8 + 16];
#pragma unroll
        for (int t = 0; t < 4; ++t) {
            AFrag b;
            const __bf16* wRow =
                &sW2[(t * 16 + nlo) * ZS + kb * 32 + (half ? 8 : 0)];
            b.q[0] = *(const uint4*)&wRow[0];
            b.q[1] = *(const uint4*)&wRow[16];
            accv[t] = __builtin_amdgcn_wmma_f32_16x16x32_bf16(
                false, a.v, false, b.v, (short)0, accv[t], false, false);
        }
    }

#pragma unroll
    for (int t = 0; t < 4; ++t) {
#pragma unroll
        for (int v = 0; v < 8; ++v) {
            int m = v + base8;
            if (m < cnt) {
                float val = relu_f(accv[t][v] + b2v[t]);
                out[(size_t)(n0 + m) * Dv + t * 16 + nlo] = val;
            }
        }
    }
}

// ---------------------------------------------------------------------------
// Host launcher
// ---------------------------------------------------------------------------
extern "C" void kernel_launch(void* const* d_in, const int* in_sizes, int n_in,
                              void* d_out, int out_size, void* d_ws,
                              size_t ws_size, hipStream_t stream) {
    const int N_ = in_sizes[0] / Dv;       // 50000
    const int E_ = in_sizes[1] / 2;        // 300000

    const float* x[3];
    const int *upI[3], *dnI[3];
    const float *upA[3], *dnA[3];
    for (int d = 0; d < 3; ++d) {
        x[d]   = (const float*)d_in[d * 5 + 0];
        upI[d] = (const int*)  d_in[d * 5 + 1];
        dnI[d] = (const int*)  d_in[d * 5 + 2];
        upA[d] = (const float*)d_in[d * 5 + 3];
        dnA[d] = (const float*)d_in[d * 5 + 4];
    }
    const float* Wup = (const float*)d_in[15];
    const float* bup = (const float*)d_in[16];
    const float* Wdn = (const float*)d_in[17];
    const float* bdn = (const float*)d_in[18];
    const float* W1  = (const float*)d_in[19];
    const float* b1  = (const float*)d_in[20];
    const float* W2  = (const float*)d_in[21];
    const float* b2  = (const float*)d_in[22];

    // Workspace layout (all 256B aligned)
    char* ws = (char*)d_ws;
    size_t off = 0;
    auto alloc = [&](size_t bytes) {
        char* p = ws + off;
        off = (off + bytes + 255) & ~(size_t)255;
        return p;
    };
    float*  acc  = (float*) alloc((size_t)N_ * Dv * sizeof(float));
    __bf16* xbf  = (__bf16*)alloc((size_t)N_ * Dv * sizeof(__bf16));
    __bf16* WupT = (__bf16*)alloc((size_t)Dv * K2 * sizeof(__bf16));
    __bf16* WdnT = (__bf16*)alloc((size_t)Dv * K2 * sizeof(__bf16));
    __bf16* W1T  = (__bf16*)alloc((size_t)Dv * Dv * sizeof(__bf16));
    __bf16* W2T  = (__bf16*)alloc((size_t)Dv * Dv * sizeof(__bf16));
    float* out = (float*)d_out;

    prep_weights_kernel<<<(K2 * Dv + 255) / 256, 256, 0, stream>>>(
        Wup, Wdn, W1, W2, WupT, WdnT, W1T, W2T);

    const int nElem = N_ * Dv;
    for (int d = 0; d < 3; ++d) {
        prep_level_kernel<<<(nElem + 255) / 256, 256, 0, stream>>>(
            x[d], xbf, acc, nElem);
        edge_mlp_kernel<<<(E_ + 63) / 64, 128, 0, stream>>>(
            xbf, upI[d], upA[d], WupT, bup, acc, E_);
        edge_mlp_kernel<<<(E_ + 63) / 64, 128, 0, stream>>>(
            xbf, dnI[d], dnA[d], WdnT, bdn, acc, E_);
        node_mlp_kernel<<<(N_ + 63) / 64, 128, 0, stream>>>(
            acc, W1T, b1, W2T, b2, out + (size_t)d * nElem, N_);
    }
}
